// SequentialGPTOSSMoE_43679817401057
// MI455X (gfx1250) — compile-verified, matrix operand order checked by reading, not run on hardware
//
#include <hip/hip_runtime.h>
#include <cstdint>
#include <cstddef>

// ---------------- problem constants (from reference) ----------------
constexpr int   kN = 1024;          // B*T tokens
constexpr int   kH = 1440;          // hidden dim
constexpr int   kI = 1440;          // intermediate dim
constexpr int   kE = 16;            // experts
constexpr float kAlpha = 1.702f;
constexpr float kLimit = 7.0f;

// ---------------- tiling ----------------
constexpr int kTPB   = 192;         // 6 waves of 32
constexpr int kJC    = 96;          // columns staged per j-block (6 waves * 16)
constexpr int kNJ    = 15;          // 1440 / 96
constexpr int kXS    = kH + 8;      // LDS row stride in bf16 elems (bank-conflict pad)
constexpr int kWS    = 36;          // staged weight row stride in fp32 elems
constexpr int kWSz   = kJC * kWS;   // floats per staging buffer (3456)

// ---------------- dynamic LDS layout (bytes) ----------------
constexpr int kOffXs  = 0;
constexpr int kOffAct = 16 * kXS * 2;            // 46336
constexpr int kOffW   = kOffAct + 16 * kXS * 2;  // 92672 ; 4 buffers of 13824
constexpr int kWB     = kWSz * 4;                // 13824 per staging buffer
constexpr int kOffS   = kOffW + 4 * kWB;         // 147968
constexpr int kSmemBytes = kOffS + 64;           // 148032

typedef __bf16 bf16_t;
typedef __attribute__((ext_vector_type(16))) __bf16 v16bf;
typedef __attribute__((ext_vector_type(8)))  __bf16 v8bf;
typedef __attribute__((ext_vector_type(8)))  float  v8f;
typedef __attribute__((ext_vector_type(4)))  int    v4i;

#define AS1 __attribute__((address_space(1)))
#define AS3 __attribute__((address_space(3)))

#if defined(__HIP_DEVICE_COMPILE__) && \
    __has_builtin(__builtin_amdgcn_global_load_async_to_lds_b32) && \
    __has_builtin(__builtin_amdgcn_global_load_async_to_lds_b128)
#define HAVE_ASYNC_LDS 1
#endif

// Wait until the wave's ASYNCcnt drops to <= N: with double buffering we pass
// the size of the just-issued (next) batch so only the previous batch drains
// while the new one stays in flight behind the WMMAs.
template <int N>
static __device__ __forceinline__ void async_wait_n() {
#ifdef HAVE_ASYNC_LDS
#if __has_builtin(__builtin_amdgcn_s_wait_asynccnt)
  __builtin_amdgcn_s_wait_asynccnt(N);
#else
  asm volatile("s_wait_asynccnt %0" ::"i"(N) : "memory");
#endif
#endif
}

// Stage a [32 x 96] fp32 chunk of W (global row stride = ld) into LDS,
// TRANSPOSED as tw[col * kWS + row]. Coalesced global reads; the transpose
// scatter is done by the async-to-LDS engine (per-lane LDS address).
// Issues 16 async instructions per wave (ASYNCcnt += 16).
static __device__ __forceinline__ void stage_w32x96(const float* __restrict__ gsrc,
                                                    int ld, float* tw, int tid) {
#pragma unroll
  for (int s = 0; s < 16; ++s) {
    int idx = tid + s * kTPB;          // 0..3071
    int r = idx / kJC;                 // K row 0..31
    int c = idx - r * kJC;             // col 0..95
    const float* g = gsrc + (size_t)r * ld + c;
    float* l = tw + c * kWS + r;
#ifdef HAVE_ASYNC_LDS
    __builtin_amdgcn_global_load_async_to_lds_b32(
        (AS1 int*)(uintptr_t)g, (AS3 int*)(uintptr_t)l, 0, 0);
#else
    *l = *g;
#endif
  }
}

// A fragment (16x32 bf16, M x K) from an LDS row-major tile with stride kXS.
// Lane L: M = L&15, kb = (L>=16)?8:0 ; elems 0..7 = K kb..kb+7, 8..15 = K kb+16..kb+23.
static __device__ __forceinline__ v16bf load_a_frag(const bf16_t* tile, int lane, int k0) {
  int M  = lane & 15;
  int kb = (lane >> 4) << 3;
  const bf16_t* p = tile + M * kXS + k0 + kb;
  v8bf lo = *(const v8bf*)(p);
  v8bf hi = *(const v8bf*)(p + 16);
  v16bf a;
#pragma unroll
  for (int i = 0; i < 8; ++i) { a[i] = lo[i]; a[i + 8] = hi[i]; }
  return a;
}

// B fragment (32x16 bf16, K x N) from the transposed fp32 staging buffer.
// Lane L holds column N = lc + (L&15); same K packing as A.
static __device__ __forceinline__ v16bf load_b_frag(const float* tw, int lane, int lc) {
  int n  = lane & 15;
  int kb = (lane >> 4) << 3;
  const float* p = tw + (lc + n) * kWS + kb;
  float4 f0 = *(const float4*)(p);
  float4 f1 = *(const float4*)(p + 4);
  float4 f2 = *(const float4*)(p + 16);
  float4 f3 = *(const float4*)(p + 20);
  v16bf b;
  b[0]  = (bf16_t)f0.x; b[1]  = (bf16_t)f0.y; b[2]  = (bf16_t)f0.z; b[3]  = (bf16_t)f0.w;
  b[4]  = (bf16_t)f1.x; b[5]  = (bf16_t)f1.y; b[6]  = (bf16_t)f1.z; b[7]  = (bf16_t)f1.w;
  b[8]  = (bf16_t)f2.x; b[9]  = (bf16_t)f2.y; b[10] = (bf16_t)f2.z; b[11] = (bf16_t)f2.w;
  b[12] = (bf16_t)f3.x; b[13] = (bf16_t)f3.y; b[14] = (bf16_t)f3.z; b[15] = (bf16_t)f3.w;
  return b;
}

// One K=32 step of the dual (gate, up) GEMM.
static __device__ __forceinline__ void gemm_step2(const bf16_t* xs, const float* twg,
                                                  const float* twu, int lane, int wv,
                                                  int k0, v8f& accG, v8f& accU) {
  v16bf a  = load_a_frag(xs, lane, k0);
  v16bf bg = load_b_frag(twg, lane, wv * 16);
  v16bf bu = load_b_frag(twu, lane, wv * 16);
  accG = __builtin_amdgcn_wmma_f32_16x16x32_bf16(false, a, false, bg, (short)0, accG, false, false);
  accU = __builtin_amdgcn_wmma_f32_16x16x32_bf16(false, a, false, bu, (short)0, accU, false, false);
}

// One K=32 step of the down-projection GEMM.
static __device__ __forceinline__ void gemm_step1(const bf16_t* act, const float* twd,
                                                  int lane, int wv, int k0, v8f& acc) {
  v16bf a  = load_a_frag(act, lane, k0);
  v16bf bd = load_b_frag(twd, lane, wv * 16);
  acc = __builtin_amdgcn_wmma_f32_16x16x32_bf16(false, a, false, bd, (short)0, acc, false, false);
}

// ---------------- router: logits -> top4 softmax scores + x->bf16 ----------------
__global__ __launch_bounds__(32) void router_kernel(const float* __restrict__ hs,
                                                    const float* __restrict__ rw,
                                                    const float* __restrict__ rb,
                                                    float* __restrict__ scores,
                                                    bf16_t* __restrict__ xb) {
  const int n = blockIdx.x;
  const int lane = threadIdx.x;
  const float* x = hs + (size_t)n * kH;

  float logit[kE];
#pragma unroll
  for (int e = 0; e < kE; ++e) {
    const float* w = rw + (size_t)e * kH;
    float p = 0.f;
    for (int h = lane; h < kH; h += 32) p += x[h] * w[h];
#pragma unroll
    for (int off = 16; off > 0; off >>= 1) p += __shfl_xor(p, off, 32);
    logit[e] = p + rb[e];
  }

  // convert this token's activations to bf16 for the WMMA pipeline
  for (int h = lane; h < kH; h += 32) xb[(size_t)n * kH + h] = (bf16_t)x[h];

  // top-4 (all lanes redundantly; no divergence)
  float tmp[kE];
#pragma unroll
  for (int e = 0; e < kE; ++e) tmp[e] = logit[e];
  float val[4]; int idx[4];
#pragma unroll
  for (int k = 0; k < 4; ++k) {
    float best = -3.4e38f; int bi = 0;
#pragma unroll
    for (int e = 0; e < kE; ++e)
      if (tmp[e] > best) { best = tmp[e]; bi = e; }
    val[k] = best; idx[k] = bi; tmp[bi] = -3.4e38f;
  }
  float mx = val[0];
  float p0 = __expf(val[0] - mx), p1 = __expf(val[1] - mx);
  float p2 = __expf(val[2] - mx), p3 = __expf(val[3] - mx);
  float inv = 1.0f / (p0 + p1 + p2 + p3);

  if (lane < kE) {
    float sc = 0.f;
    if      (idx[0] == lane) sc = p0 * inv;
    else if (idx[1] == lane) sc = p1 * inv;
    else if (idx[2] == lane) sc = p2 * inv;
    else if (idx[3] == lane) sc = p3 * inv;
    scores[(size_t)n * kE + lane] = sc;
  }
}

// ---------------- fused expert MLP: one (expert, 16-token tile) per block ----------------
__global__ __launch_bounds__(kTPB, 1) void moe_mlp_kernel(
    const bf16_t* __restrict__ xb,
    const float* __restrict__ gate_w, const float* __restrict__ gate_b,
    const float* __restrict__ up_w,   const float* __restrict__ up_b,
    const float* __restrict__ down_w, const float* __restrict__ down_b,
    const float* __restrict__ scores, float* __restrict__ out) {
  extern __shared__ char smem[];
  bf16_t* xs   = (bf16_t*)(smem + kOffXs);   // 16 x kXS bf16 : X tile
  bf16_t* act  = (bf16_t*)(smem + kOffAct);  // 16 x kXS bf16 : activation tile
  float* tws   = (float*)(smem + kOffW);     // 4 staging buffers of kWSz floats
  float* stile = (float*)(smem + kOffS);     // 16 scores

  const int tile = blockIdx.x;     // token tile 0..63
  const int e    = blockIdx.y;     // expert 0..15
  const int m0   = tile * 16;
  const int tid  = threadIdx.x;
  const int lane = tid & 31;
  const int wv   = tid >> 5;       // 0..5

  if (tid < 16) stile[tid] = scores[(size_t)(m0 + tid) * kE + e];
  __syncthreads();
  float ssum = 0.f;
#pragma unroll
  for (int m = 0; m < 16; ++m) ssum += stile[m];
  if (ssum == 0.0f) return;        // no token in this tile routed to expert e

  // X tile -> LDS (16 rows x 1440 bf16), 16B chunks, async if available
  for (int c = tid; c < 16 * 180; c += kTPB) {
    int row = c / 180, cc = c - row * 180;
    const bf16_t* g = xb + (size_t)(m0 + row) * kH + cc * 8;
    bf16_t* l = xs + row * kXS + cc * 8;
#ifdef HAVE_ASYNC_LDS
    __builtin_amdgcn_global_load_async_to_lds_b128(
        (AS1 v4i*)(uintptr_t)g, (AS3 v4i*)(uintptr_t)l, 0, 0);
#else
    *(v8bf*)l = *(const v8bf*)g;
#endif
  }
  async_wait_n<0>();
  __syncthreads();

  const float* Wg = gate_w + (size_t)e * kH * kI;
  const float* Wu = up_w   + (size_t)e * kH * kI;

  // ---- phase 1: gate/up GEMMs + GLU, act tile into LDS ----
  // Double-buffered: stage k+1 into pong, wait only for ping's 32 async ops.
  for (int j = 0; j < kNJ; ++j) {
    const int jc0 = j * kJC;
    v8f accG = {}; v8f accU = {};
    stage_w32x96(Wg + jc0, kI, tws, tid);
    stage_w32x96(Wu + jc0, kI, tws + kWSz, tid);
    int pp = 0;
#pragma unroll 1
    for (int k0 = 0; k0 < kH - 32; k0 += 32) {
      const int np = pp ^ 1;
      stage_w32x96(Wg + (size_t)(k0 + 32) * kI + jc0, kI, tws + (np * 2 + 0) * kWSz, tid);
      stage_w32x96(Wu + (size_t)(k0 + 32) * kI + jc0, kI, tws + (np * 2 + 1) * kWSz, tid);
      async_wait_n<32>();   // drain ping batch; pong batch stays in flight
      __syncthreads();
      gemm_step2(xs, tws + (pp * 2 + 0) * kWSz, tws + (pp * 2 + 1) * kWSz,
                 lane, wv, k0, accG, accU);
      __syncthreads();
      pp = np;
    }
    async_wait_n<0>();      // final batch
    __syncthreads();
    gemm_step2(xs, tws + (pp * 2 + 0) * kWSz, tws + (pp * 2 + 1) * kWSz,
               lane, wv, kH - 32, accG, accU);
    __syncthreads();        // staging buffers free for next j

    const int i    = jc0 + wv * 16 + (lane & 15);
    const int mrow = (lane >> 4) << 3;
    const float gb = gate_b[(size_t)e * kI + i];
    const float ub = up_b[(size_t)e * kI + i];
#pragma unroll
    for (int r = 0; r < 8; ++r) {
      float g = accG[r] + gb;
      float u = accU[r] + ub;
      g = fminf(g, kLimit);
      u = fminf(fmaxf(u, -kLimit), kLimit);
      float glu = g * __builtin_amdgcn_rcpf(1.0f + __expf(-kAlpha * g));
      act[(mrow + r) * kXS + i] = (bf16_t)((u + 1.0f) * glu);
    }
  }
  __syncthreads();          // act tile complete & visible block-wide

  // ---- phase 2: down projection + score-weighted combine ----
  const float* Wd = down_w + (size_t)e * kI * kH;
  for (int j = 0; j < kNJ; ++j) {
    const int jc0 = j * kJC;
    v8f acc = {};
    stage_w32x96(Wd + jc0, kH, tws, tid);
    int pp = 0;
#pragma unroll 1
    for (int k0 = 0; k0 < kI - 32; k0 += 32) {
      const int np = pp ^ 1;
      stage_w32x96(Wd + (size_t)(k0 + 32) * kH + jc0, kH, tws + np * kWSz, tid);
      async_wait_n<16>();
      __syncthreads();
      gemm_step1(act, tws + pp * kWSz, lane, wv, k0, acc);
      __syncthreads();
      pp = np;
    }
    async_wait_n<0>();
    __syncthreads();
    gemm_step1(act, tws + pp * kWSz, lane, wv, kI - 32, acc);
    __syncthreads();

    const int h    = jc0 + wv * 16 + (lane & 15);
    const int mrow = (lane >> 4) << 3;
    const float db = down_b[(size_t)e * kH + h];
#pragma unroll
    for (int r = 0; r < 8; ++r) {
      float s = stile[mrow + r];
      if (s != 0.0f) {
        atomicAdd(&out[(size_t)(m0 + mrow + r) * kH + h], s * (acc[r] + db));
      }
    }
  }
}

extern "C" void kernel_launch(void* const* d_in, const int* in_sizes, int n_in,
                              void* d_out, int out_size, void* d_ws, size_t ws_size,
                              hipStream_t stream) {
  (void)in_sizes; (void)n_in; (void)out_size; (void)ws_size;
  const float* hs       = (const float*)d_in[0];
  const float* router_w = (const float*)d_in[1];
  const float* router_b = (const float*)d_in[2];
  const float* gate_w   = (const float*)d_in[3];
  const float* gate_b   = (const float*)d_in[4];
  const float* up_w     = (const float*)d_in[5];
  const float* up_b     = (const float*)d_in[6];
  const float* down_w   = (const float*)d_in[7];
  const float* down_b   = (const float*)d_in[8];

  float* out        = (float*)d_out;
  float* out_scores = out + (size_t)kN * kH;   // second tuple output
  bf16_t* xb        = (bf16_t*)d_ws;           // 1024*1440 bf16 (~2.9 MB)

  (void)hipFuncSetAttribute(reinterpret_cast<const void*>(moe_mlp_kernel),
                            hipFuncAttributeMaxDynamicSharedMemorySize, kSmemBytes);

  (void)hipMemsetAsync(out, 0, (size_t)kN * kH * sizeof(float), stream);

  router_kernel<<<kN, 32, 0, stream>>>(hs, router_w, router_b, out_scores, xb);

  dim3 grid(kN / 16, kE, 1);
  moe_mlp_kernel<<<grid, kTPB, kSmemBytes, stream>>>(
      xb, gate_w, gate_b, up_w, up_b, down_w, down_b, out_scores, out);
}
